// SAttention_20212116095048
// MI455X (gfx1250) — matrix operand order repaired and checked by reading, not compiled, not run on hardware
//
#include <hip/hip_runtime.h>
#include <hip/hip_bf16.h>

// ---------------- types ----------------
typedef __attribute__((ext_vector_type(16))) __bf16 v16bf;
typedef __attribute__((ext_vector_type(8)))  __bf16 v8bf;
typedef __attribute__((ext_vector_type(8)))  float  v8f;
typedef __attribute__((ext_vector_type(4)))  unsigned int v4u;
typedef __attribute__((ext_vector_type(8)))  int v8i;
typedef __attribute__((ext_vector_type(4)))  int v4i;

#define S_LEN  512
#define B_SZ   64
#define D_MODEL 512
#define NHEAD  8
#define DH     64
#define NTOK   (S_LEN * B_SZ)       // 32768
#define INV_TEMP 0.125f             // 1/sqrt(D/H) = 1/8

#if __has_builtin(__builtin_amdgcn_tensor_load_to_lds)
#  define HAVE_TDM 1
#  if __has_include(<hip/amd_detail/amd_gfx1250_TDM.h>)
#    define HAVE_TDM6 1   /* therock headers => 6-arg builtin */
#  endif
#endif

// round-to-nearest-even f32 -> bf16
static __device__ __forceinline__ __bf16 f2bf(float f) {
    unsigned u = __float_as_uint(f);
    u += 0x7fffu + ((u >> 16) & 1u);
    unsigned short h = (unsigned short)(u >> 16);
    __bf16 r;
    __builtin_memcpy(&r, &h, 2);
    return r;
}

static __device__ __forceinline__ v16bf make_afrag(const __bf16* lo_p, const __bf16* hi_p) {
    v8bf lo = *(const v8bf*)lo_p;
    v8bf hi = *(const v8bf*)(lo_p + 8);
    v16bf r;
#pragma unroll
    for (int i = 0; i < 8; ++i) { r[i] = lo[i]; r[i + 8] = hi[i]; }
    return r;
}

static __device__ __forceinline__ v8f wmma_bf16(v16bf a, v16bf b, v8f c) {
    return __builtin_amdgcn_wmma_f32_16x16x32_bf16(false, a, false, b, (short)0, c, false, false);
}

// ---------------- TDM: stage nrows x 512 bf16 rows into LDS ----------------
#if defined(HAVE_TDM)
static __device__ __forceinline__ void tdm_rows_to_lds(const __bf16* gsrc, void* lds_dst,
                                                       unsigned nrows) {
    unsigned long long ga = (unsigned long long)(uintptr_t)gsrc;
    unsigned int lo = (unsigned int)(uintptr_t)lds_dst;
    v4u g0;
    g0[0] = 1u;                                        // count=1, no gather
    g0[1] = lo;                                        // lds_addr (bytes)
    g0[2] = (unsigned)ga;                              // global_addr[31:0]
    g0[3] = (unsigned)(ga >> 32) | (2u << 30);         // global_addr[56:32] | type=2
    v8i g1;
    g1[0] = (int)(1u << 16);                           // data_size=2B
    g1[1] = (int)((unsigned)D_MODEL << 16);            // tensor_dim0[15:0]=512
    g1[2] = (int)((unsigned)D_MODEL << 16);            // tensor_dim1[15:0]=512
    g1[3] = (int)((unsigned)D_MODEL << 16);            // tile_dim0=512
    g1[4] = (int)nrows;                                // tile_dim1=nrows, tile_dim2=0
    g1[5] = D_MODEL;                                   // tensor_dim0_stride low = 512
    g1[6] = 0;
    g1[7] = 0;
    v4i z = {0, 0, 0, 0};
#if defined(HAVE_TDM6)
    v8i z8 = {0, 0, 0, 0, 0, 0, 0, 0};
    __builtin_amdgcn_tensor_load_to_lds(g0, g1, z, z, z8, 0);
#else
    __builtin_amdgcn_tensor_load_to_lds(g0, g1, z, z, 0);
#endif
    __builtin_amdgcn_s_wait_tensorcnt(0);
}
#endif

// stage 64 rows (64 output cols worth) of a [512,512] bf16 weight into LDS
static __device__ __forceinline__ void stage_weights(const __bf16* __restrict__ W, int n0,
                                                     __bf16* wl) {
#if defined(HAVE_TDM)
    if (threadIdx.x < 32) {
        tdm_rows_to_lds(W + (size_t)n0 * D_MODEL, wl, 64u);
    }
    __syncthreads();
#else
    const uint4* src = (const uint4*)(W + (size_t)n0 * D_MODEL);
    uint4* dst = (uint4*)wl;
    for (int i = threadIdx.x; i < (64 * D_MODEL) / 8; i += 256) dst[i] = src[i];
    __syncthreads();
#endif
}

// ---------------- weight fp32 -> bf16 ----------------
__global__ void cvt_w_kernel(const float* wq, const float* wk, const float* wv,
                             const float* w1, const float* w2, __bf16* out) {
    int i = blockIdx.x * blockDim.x + threadIdx.x;   // 5*262144 total
    const int per = D_MODEL * D_MODEL;
    int m = i / per, r = i - m * per;
    const float* src = (m == 0) ? wq : (m == 1) ? wk : (m == 2) ? wv : (m == 3) ? w1 : w2;
    out[i] = f2bf(src[r]);
}

// ---------------- LayerNorm (+optional residual) ----------------
__global__ void ln_kernel(const float* __restrict__ in, const float* __restrict__ res,
                          const float* __restrict__ g, const float* __restrict__ be,
                          float* __restrict__ outf, __bf16* __restrict__ outb) {
    __shared__ float red[256];
    const int row = blockIdx.x;
    const int tid = threadIdx.x;
    const size_t base = (size_t)row * D_MODEL;
    float v0 = in[base + tid];
    float v1 = in[base + 256 + tid];
    if (res) { v0 += res[base + tid]; v1 += res[base + 256 + tid]; }

    red[tid] = v0 + v1;
    __syncthreads();
    for (int s = 128; s > 0; s >>= 1) { if (tid < s) red[tid] += red[tid + s]; __syncthreads(); }
    const float mu = red[0] * (1.0f / D_MODEL);
    __syncthreads();

    const float d0 = v0 - mu, d1 = v1 - mu;
    red[tid] = d0 * d0 + d1 * d1;
    __syncthreads();
    for (int s = 128; s > 0; s >>= 1) { if (tid < s) red[tid] += red[tid + s]; __syncthreads(); }
    const float rstd = rsqrtf(red[0] * (1.0f / D_MODEL) + 1e-5f);

    const float y0 = d0 * rstd * g[tid] + be[tid];
    const float y1 = d1 * rstd * g[tid + 256] + be[tid + 256];
    outf[base + tid]       = y0;
    outf[base + 256 + tid] = y1;
    outb[base + tid]       = f2bf(y0);
    outb[base + 256 + tid] = f2bf(y1);
}

// ---------------- GEMM core: 32x64 tile per wave, B from LDS, A pipelined ----------------
static __device__ __forceinline__ void gemm_core(const __bf16* __restrict__ act,
                                                 const __bf16* wl, int m0, int lane,
                                                 v8f (&c)[2][4]) {
    const int row = lane & 15;
    const int hi  = lane >> 4;
    const int kbA = hi * 16;
    const int kbB = hi * 16;
    const __bf16* ar0 = act + (size_t)(m0 + row) * D_MODEL + kbA;
    const __bf16* ar1 = ar0 + (size_t)16 * D_MODEL;
    const __bf16* bb  = wl + (size_t)row * D_MODEL + kbB;   // row == lane&15 == B col

    v16bf a0 = make_afrag(ar0, ar0 + 16);
    v16bf a1 = make_afrag(ar1, ar1 + 16);
#pragma unroll 4
    for (int k0 = 0; k0 < D_MODEL; k0 += 32) {
        v16bf na0 = a0, na1 = a1;
        if (k0 + 32 < D_MODEL) {                    // prefetch next A fragments
            na0 = make_afrag(ar0 + k0 + 32, ar0 + k0 + 48);
            na1 = make_afrag(ar1 + k0 + 32, ar1 + k0 + 48);
        }
#pragma unroll
        for (int j = 0; j < 4; ++j) {
            v16bf b = *(const v16bf*)(bb + (size_t)j * 16 * D_MODEL + k0);  // LDS
            c[0][j] = wmma_bf16(a0, b, c[0][j]);
            c[1][j] = wmma_bf16(a1, b, c[1][j]);
        }
        a0 = na0; a1 = na1;
    }
}

// ---------------- fused QKV projection ----------------
// grid: 128 mblocks * 24 nblocks; block: 256 threads (8 waves), each wave 32x64
__global__ __launch_bounds__(256)
void qkv_kernel(const __bf16* __restrict__ act,
                const __bf16* __restrict__ wq, const __bf16* __restrict__ wk,
                const __bf16* __restrict__ wv,
                __bf16* __restrict__ q, __bf16* __restrict__ kk, __bf16* __restrict__ vt) {
    __shared__ __bf16 wl[64 * D_MODEL];            // 64 KB weight tile
    const int lane = threadIdx.x & 31;
    const int wave = threadIdx.x >> 5;
    const int mblock = blockIdx.x & 127;
    const int nblk   = blockIdx.x >> 7;            // 0..23
    const int m0  = (mblock * 8 + wave) * 32;
    const int n0g = nblk * 64;                     // global col in [0,1536)
    const int mat = n0g >> 9;                      // 0:q 1:k 2:v
    const int n0  = n0g & 511;
    const __bf16* W = (mat == 0) ? wq : (mat == 1) ? wk : wv;

    stage_weights(W, n0, wl);

    v8f c[2][4] = {};
    gemm_core(act, wl, m0, lane, c);

    const int lo = lane & 15, hi = lane >> 4;
#pragma unroll
    for (int mm = 0; mm < 2; ++mm) {
#pragma unroll
        for (int r = 0; r < 8; ++r) {
            const int tok = m0 + mm * 16 + r + 8 * hi;
            const int s = tok >> 6, b = tok & 63;
#pragma unroll
            for (int j = 0; j < 4; ++j) {
                const int col = n0 + j * 16 + lo;
                const int h = col >> 6, dh = col & 63;
                const __bf16 val = f2bf(c[mm][j][r]);
                if (mat == 2)
                    vt[(((size_t)(b * NHEAD + h) * DH + dh) * S_LEN) + s] = val;
                else {
                    __bf16* dst = (mat == 0) ? q : kk;
                    dst[(((size_t)(b * NHEAD + h) * S_LEN + s) * DH) + dh] = val;
                }
            }
        }
    }
}

// ---------------- attention: one wave per (b,h, 16 query rows) ----------------
__global__ __launch_bounds__(32)
void attn_kernel(const __bf16* __restrict__ q, const __bf16* __restrict__ k,
                 const __bf16* __restrict__ vt, float* __restrict__ att) {
    __shared__ float  sc[16 * S_LEN];   // 32 KB
    __shared__ __bf16 pr[16 * S_LEN];   // 16 KB
    const int lane = threadIdx.x;
    const int qt = blockIdx.x & 31;
    const int bh = blockIdx.x >> 5;     // b*8 + h
    const int s0 = qt * 16;
    const __bf16* qp = q  + ((size_t)bh * S_LEN + s0) * DH;
    const __bf16* kp = k  + (size_t)bh * S_LEN * DH;
    const __bf16* vp = vt + (size_t)bh * DH * S_LEN;
    const int row = lane & 15, hi = lane >> 4;
    const int kbA = hi * 16, kbB = hi * 16;

    // --- scores = Q K^T * inv_temp (B prefetched one tile ahead) ---
    const __bf16* qrow = qp + row * DH;
    v16bf a0 = make_afrag(qrow + kbA,      qrow + 16 + kbA);
    v16bf a1 = make_afrag(qrow + 32 + kbA, qrow + 48 + kbA);
    const __bf16* kbase = kp + row * DH + kbB;   // B col = lane&15 == row
    v16bf b0 = *(const v16bf*)(kbase);
    v16bf b1 = *(const v16bf*)(kbase + 32);
    for (int tt = 0; tt < 32; ++tt) {
        v16bf nb0 = b0, nb1 = b1;
        if (tt < 31) {
            nb0 = *(const v16bf*)(kbase + (tt + 1) * 16 * DH);
            nb1 = *(const v16bf*)(kbase + (tt + 1) * 16 * DH + 32);
        }
        v8f c = {};
        c = wmma_bf16(a0, b0, c);
        c = wmma_bf16(a1, b1, c);
#pragma unroll
        for (int r = 0; r < 8; ++r)
            sc[(r + 8 * hi) * S_LEN + tt * 16 + row] = c[r] * INV_TEMP;
        b0 = nb0; b1 = nb1;
    }
    __syncthreads();

    // --- softmax over t (512) : 2 lanes per row ---
    {
        const int rr = lane >> 1;
        const int c0 = (lane & 1) * 256;
        float* srow = sc + rr * S_LEN + c0;
        const float4* s4 = (const float4*)srow;
        float m = -3.4e38f;
        for (int i = 0; i < 64; ++i) {
            float4 v = s4[i];
            m = fmaxf(m, fmaxf(fmaxf(v.x, v.y), fmaxf(v.z, v.w)));
        }
        m = fmaxf(m, __shfl_xor(m, 1, 32));
        float sum = 0.f;
        float4* s4w = (float4*)srow;
        for (int i = 0; i < 64; ++i) {
            float4 v = s4w[i];
            v.x = __expf(v.x - m); v.y = __expf(v.y - m);
            v.z = __expf(v.z - m); v.w = __expf(v.w - m);
            sum += v.x + v.y + v.z + v.w;
            s4w[i] = v;
        }
        sum += __shfl_xor(sum, 1, 32);
        const float rs = 1.0f / sum;
        __bf16* prow = pr + rr * S_LEN + c0;
        for (int i = 0; i < 64; ++i) {
            float4 v = s4w[i];
            prow[4 * i + 0] = f2bf(v.x * rs);
            prow[4 * i + 1] = f2bf(v.y * rs);
            prow[4 * i + 2] = f2bf(v.z * rs);
            prow[4 * i + 3] = f2bf(v.w * rs);
        }
    }
    __syncthreads();

    // --- out = P @ V  (V transposed; B prefetched one k-step ahead) ---
    v8f c[4] = {};
    const __bf16* prA = pr + row * S_LEN;
    const __bf16* vb0 = vp + row * S_LEN + kbB;
    v16bf bv[4];
#pragma unroll
    for (int j = 0; j < 4; ++j) bv[j] = *(const v16bf*)(vb0 + (size_t)j * 16 * S_LEN);
    for (int ks = 0; ks < 16; ++ks) {
        v16bf nbv[4];
#pragma unroll
        for (int j = 0; j < 4; ++j) nbv[j] = bv[j];
        if (ks < 15) {
#pragma unroll
            for (int j = 0; j < 4; ++j)
                nbv[j] = *(const v16bf*)(vb0 + (size_t)j * 16 * S_LEN + (ks + 1) * 32);
        }
        v16bf a = make_afrag(prA + ks * 32 + kbA, prA + ks * 32 + 16 + kbA);
#pragma unroll
        for (int j = 0; j < 4; ++j) c[j] = wmma_bf16(a, bv[j], c[j]);
#pragma unroll
        for (int j = 0; j < 4; ++j) bv[j] = nbv[j];
    }
    const int bb = bh >> 3, hh = bh & 7;
#pragma unroll
    for (int r = 0; r < 8; ++r) {
        const int s = s0 + r + 8 * hi;
#pragma unroll
        for (int j = 0; j < 4; ++j) {
            const int d = hh * DH + j * 16 + row;
            att[((size_t)s * B_SZ + bb) * D_MODEL + d] = c[j][r];
        }
    }
}

// ---------------- FFN1: relu(x W1^T + b1) -> bf16 ----------------
__global__ __launch_bounds__(256)
void ffn1_kernel(const __bf16* __restrict__ act, const __bf16* __restrict__ w1,
                 const float* __restrict__ bias, __bf16* __restrict__ h1) {
    __shared__ __bf16 wl[64 * D_MODEL];
    const int lane = threadIdx.x & 31;
    const int wave = threadIdx.x >> 5;
    const int m0 = ((blockIdx.x & 127) * 8 + wave) * 32;
    const int n0 = (blockIdx.x >> 7) * 64;

    stage_weights(w1, n0, wl);
    v8f c[2][4] = {};
    gemm_core(act, wl, m0, lane, c);

    const int lo = lane & 15, hi = lane >> 4;
#pragma unroll
    for (int mm = 0; mm < 2; ++mm)
#pragma unroll
        for (int r = 0; r < 8; ++r) {
            const size_t tok = m0 + mm * 16 + r + 8 * hi;
#pragma unroll
            for (int j = 0; j < 4; ++j) {
                const int col = n0 + j * 16 + lo;
                float v = c[mm][j][r] + bias[col];
                h1[tok * D_MODEL + col] = f2bf(fmaxf(v, 0.0f));
            }
        }
}

// ---------------- FFN2: xt + h1 W2^T + b2 -> fp32 out ----------------
__global__ __launch_bounds__(256)
void ffn2_kernel(const __bf16* __restrict__ act, const __bf16* __restrict__ w2,
                 const float* __restrict__ bias, const float* __restrict__ xt,
                 float* __restrict__ out) {
    __shared__ __bf16 wl[64 * D_MODEL];
    const int lane = threadIdx.x & 31;
    const int wave = threadIdx.x >> 5;
    const int m0 = ((blockIdx.x & 127) * 8 + wave) * 32;
    const int n0 = (blockIdx.x >> 7) * 64;

    stage_weights(w2, n0, wl);
    v8f c[2][4] = {};
    gemm_core(act, wl, m0, lane, c);

    const int lo = lane & 15, hi = lane >> 4;
#pragma unroll
    for (int mm = 0; mm < 2; ++mm)
#pragma unroll
        for (int r = 0; r < 8; ++r) {
            const size_t tok = m0 + mm * 16 + r + 8 * hi;
#pragma unroll
            for (int j = 0; j < 4; ++j) {
                const int col = n0 + j * 16 + lo;
                out[tok * D_MODEL + col] = xt[tok * D_MODEL + col] + c[mm][j][r] + bias[col];
            }
        }
}

// ---------------- host launcher ----------------
extern "C" void kernel_launch(void* const* d_in, const int* in_sizes, int n_in,
                              void* d_out, int out_size, void* d_ws, size_t ws_size,
                              hipStream_t stream) {
    const float* x   = (const float*)d_in[0];
    const float* Wq  = (const float*)d_in[1];
    const float* Wk  = (const float*)d_in[2];
    const float* Wv  = (const float*)d_in[3];
    const float* g1  = (const float*)d_in[4];
    const float* b1  = (const float*)d_in[5];
    const float* g2  = (const float*)d_in[6];
    const float* b2  = (const float*)d_in[7];
    const float* W1  = (const float*)d_in[8];
    const float* bf1 = (const float*)d_in[9];
    const float* W2  = (const float*)d_in[10];
    const float* bf2 = (const float*)d_in[11];
    float* out = (float*)d_out;

    char* ws = (char*)d_ws;
    const size_t WMAT = (size_t)D_MODEL * D_MODEL;       // 262144 elements
    __bf16* wb  = (__bf16*)ws;                           // 5 bf16 weight mats
    __bf16* wqb = wb, *wkb = wb + WMAT, *wvb = wb + 2 * WMAT;
    __bf16* w1b = wb + 3 * WMAT, *w2b = wb + 4 * WMAT;
    float*  xnf = (float*) (ws + 2621440ull);            // 64 MB
    __bf16* xnb = (__bf16*)(ws + 69730304ull);           // 32 MB (reused as xt bf16)
    __bf16* qb  = (__bf16*)(ws + 103284736ull);          // 32 MB (reused as h1)
    __bf16* kb  = (__bf16*)(ws + 136839168ull);          // 32 MB
    __bf16* vtb = (__bf16*)(ws + 170393600ull);          // 32 MB
    float*  xtf = (float*) (ws + 203948032ull);          // 64 MB

    // 1) weights -> bf16
    cvt_w_kernel<<<5120, 256, 0, stream>>>(Wq, Wk, Wv, W1, W2, wb);
    // 2) LN1
    ln_kernel<<<NTOK, 256, 0, stream>>>(x, nullptr, g1, b1, xnf, xnb);
    // 3) fused QKV projection (Q,K as [B,H,S,dh]; V transposed [B,H,dh,S])
    qkv_kernel<<<128 * 24, 256, 0, stream>>>(xnb, wqb, wkb, wvb, qb, kb, vtb);
    // 4) attention -> att scratch in d_out
    attn_kernel<<<B_SZ * NHEAD * (S_LEN / 16), 32, 0, stream>>>(qb, kb, vtb, out);
    // 5) residual + LN2
    ln_kernel<<<NTOK, 256, 0, stream>>>(xnf, out, g2, b2, xtf, xnb);
    // 6) FFN1 (relu)
    ffn1_kernel<<<128 * 8, 256, 0, stream>>>(xnb, w1b, bf1, qb);
    // 7) FFN2 + residual -> final output
    ffn2_kernel<<<128 * 8, 256, 0, stream>>>(qb, w2b, bf2, xtf, out);
}
